// MultiHeadAttention_55886114455957
// MI455X (gfx1250) — matrix operand order repaired and checked by reading, not compiled, not run on hardware
//
#include <hip/hip_runtime.h>

namespace {

constexpr int kB = 2, kS = 2048, kE = 2048, kH = 16, kD = 128;
constexpr int kM = kB * kS;
constexpr float kNeg = -1e30f;

typedef __attribute__((ext_vector_type(16))) __bf16 v16bf;
typedef __attribute__((ext_vector_type(8)))  float  v8f;

union AF {
  v16bf v;
  uint4 q[2];
};

__device__ __forceinline__ v8f wmma_bf16(v16bf a, v16bf b, v8f c) {
  return __builtin_amdgcn_wmma_f32_16x16x32_bf16(false, a, false, b,
                                                 (short)0, c, false, false);
}

// fp32 -> bf16 round-to-nearest-even (manual; storage type is ushort)
__device__ __forceinline__ unsigned short f2bf(float f) {
  unsigned u = __float_as_uint(f);
  u += 0x7FFFu + ((u >> 16) & 1u);
  return (unsigned short)(u >> 16);
}

// CDNA5 async DMA: global -> LDS, 16 bytes per lane, tracked by ASYNCcnt.
// VDST = per-lane LDS byte offset, VADDR = per-lane 64-bit global address.
__device__ __forceinline__ void async_g2l_b128(unsigned lds_byte_off,
                                               const void* gptr) {
  asm volatile("global_load_async_to_lds_b128 %0, %1, off"
               :: "v"(lds_byte_off),
                  "v"((unsigned long long)(uintptr_t)gptr)
               : "memory");
}
__device__ __forceinline__ void wait_async0() {
  asm volatile("s_wait_asynccnt 0x0" ::: "memory");
}

__global__ void cast_f32_to_bf16(const float* __restrict__ in,
                                 unsigned short* __restrict__ out, int n) {
  int i = (blockIdx.x * blockDim.x + threadIdx.x) * 8;
  if (i >= n) return;
  float4 a = *(const float4*)(in + i);
  float4 b = *(const float4*)(in + i + 4);
  ushort4 r0 = make_ushort4(f2bf(a.x), f2bf(a.y), f2bf(a.z), f2bf(a.w));
  ushort4 r1 = make_ushort4(f2bf(b.x), f2bf(b.y), f2bf(b.z), f2bf(b.w));
  *(ushort4*)(out + i) = r0;
  *(ushort4*)(out + i + 4) = r1;
}

// Kt[(b*H+h)][d][s] = K[(b*S+s)][h*D+d]   (LDS-tiled 32x32 transpose)
__global__ void transpose_k(const unsigned short* __restrict__ K,
                            unsigned short* __restrict__ Kt) {
  __shared__ unsigned short t[32][33];
  const int bh = blockIdx.z, b = bh / kH, h = bh % kH;
  const int s0 = blockIdx.x * 32, d0 = blockIdx.y * 32;
  for (int r = threadIdx.y; r < 32; r += 8)
    t[r][threadIdx.x] =
        K[(size_t)(b * kS + s0 + r) * kE + h * kD + d0 + threadIdx.x];
  __syncthreads();
  for (int r = threadIdx.y; r < 32; r += 8)
    Kt[((size_t)bh * kD + d0 + r) * kS + s0 + threadIdx.x] = t[threadIdx.x][r];
}

// ---------------------------------------------------------------------------
// C[M,N] = A[M,K] @ Bw[K,N]; row-major bf16, fp32 accumulate.
// Block = 8 waves (256 thr) tiled 2(M) x 4(N); wave tile 64x64 (4x4 WMMA acc).
// Block tile 128(M) x 256(N); per k-step of 32 the block's A (128x32) and
// B (32x256) tiles are DMA'd into double-buffered LDS with
// global_load_async_to_lds_b128 (ASYNCcnt), overlapping the WMMA of the
// previous tile. LDS row strides: A 56 elems (112B), B 264 elems (528B):
// 16B-aligned for b128 ds loads and spread across the 64 banks.
// ---------------------------------------------------------------------------
constexpr unsigned kAStrideE = 56;                 // elems per A LDS row
constexpr unsigned kBStrideE = 264;                // elems per B LDS row
constexpr unsigned kABufE = 128 * kAStrideE;       // 7168 elems / buffer
constexpr unsigned kBBufE = 32 * kBStrideE;        // 8448 elems / buffer
constexpr unsigned kBBaseE = 2 * kABufE;           // B region after 2 A bufs
constexpr unsigned kSmemE = 2 * kABufE + 2 * kBBufE;  // 31232 elems = 61 KB

template <bool F32OUT>
__global__ void __launch_bounds__(256)
gemm_bf16_wmma(const unsigned short* __restrict__ A,
               const unsigned short* __restrict__ Bw,
               unsigned short* __restrict__ Cbf, float* __restrict__ Cf,
               const float* __restrict__ bias, int M, int N, int K) {
  __shared__ alignas(16) unsigned short smem[kSmemE];

  // Escape smem so alias analysis must assume the async-DMA inline asm
  // (which carries a "memory" clobber) writes it; without this the
  // ds_load_b128 fragment reads below are folded to poison (smem has no
  // C++ stores -- it is written only by the async engine).
  {
    void* esc = (void*)smem;
    asm volatile("" : "+v"(esc) :: "memory");
  }

  const int tid = threadIdx.x;
  const int lane = tid & 31, wave = tid >> 5;
  const int lm = lane & 15, half = lane >> 4;
  const int mloc = (wave & 1) * 64, nloc = (wave >> 1) * 64;
  const int mB = blockIdx.y * 128, nB = blockIdx.x * 256;

  // --- staging addresses (per thread, 6 x b128 per k-step) ---
  // A tile 128x32: 512 chunks of 16B; thread handles chunks tid and tid+256.
  const int ar = tid >> 2, as = tid & 3;
  const unsigned short* aCur0 = A + (size_t)(mB + ar) * K + as * 8;
  const unsigned short* aCur1 = aCur0 + (size_t)64 * K;
  const unsigned aOff0 = (ar * kAStrideE + as * 8) * 2;
  const unsigned aOff1 = aOff0 + 64 * kAStrideE * 2;
  // B tile 32x256: 1024 chunks; thread handles tid + 256*j (rows +8 apart).
  const int br = tid >> 5, bs = tid & 31;
  const unsigned short* bCur = Bw + (size_t)br * N + nB + bs * 8;
  const unsigned bOff = (kBBaseE + br * kBStrideE + bs * 8) * 2;

  auto stage = [&](int buf) {
    const unsigned ab = buf * (kABufE * 2);
    const unsigned bb = buf * (kBBufE * 2);
    async_g2l_b128(aOff0 + ab, aCur0);
    async_g2l_b128(aOff1 + ab, aCur1);
#pragma unroll
    for (int j = 0; j < 4; ++j)
      async_g2l_b128(bOff + bb + j * (8 * kBStrideE * 2),
                     bCur + (size_t)(8 * j) * N);
  };

  v8f acc[4][4] = {};

  stage(0);
  wait_async0();
  __syncthreads();

#pragma unroll 1
  for (int k0 = 0; k0 < K; k0 += 32) {
    const int buf = (k0 >> 5) & 1;
    if (k0 + 32 < K) {  // DMA next tile while computing this one
      aCur0 += 32;
      aCur1 += 32;
      bCur += (size_t)32 * N;
      stage(buf ^ 1);
    }

    const unsigned aBase = buf * kABufE;
    const unsigned bBase = kBBaseE + buf * kBBufE;

    // A fragments (ISA 7.12.2 16-bit A layout): lane = row m, k chunks at
    // half*8 and 16+half*8
    AF af[4];
#pragma unroll
    for (int i = 0; i < 4; ++i) {
      const unsigned e = aBase + (mloc + i * 16 + lm) * kAStrideE + half * 8;
      af[i].q[0] = *(const uint4*)&smem[e];
      af[i].q[1] = *(const uint4*)&smem[e + 16];
    }
    // B fragments: lane holds column n=lm, k in table order (half*8, 16+half*8)
#pragma unroll
    for (int j = 0; j < 4; ++j) {
      AF bfr;
      const unsigned e = bBase + nloc + j * 16 + lm;
      unsigned short* bu = (unsigned short*)&bfr;
#pragma unroll
      for (int s = 0; s < 16; ++s)
        bu[s] = smem[e + ((s & 7) + half * 8 + (s >> 3) * 16) * kBStrideE];
#pragma unroll
      for (int i = 0; i < 4; ++i)
        acc[i][j] = wmma_bf16(af[i].v, bfr.v, acc[i][j]);
    }

    wait_async0();
    __syncthreads();
  }

  // C layout: lane holds col n=lane%16, rows half*8+e in element e
#pragma unroll
  for (int i = 0; i < 4; ++i)
#pragma unroll
    for (int j = 0; j < 4; ++j) {
      const int n = nB + nloc + j * 16 + lm;
#pragma unroll
      for (int e = 0; e < 8; ++e) {
        const int row = mB + mloc + i * 16 + half * 8 + e;
        if (F32OUT)
          Cf[(size_t)row * N + n] = acc[i][j][e] + bias[n];
        else
          Cbf[(size_t)row * N + n] = f2bf(acc[i][j][e]);
      }
    }
}

// Flash attention, causal + ALiBi (mask before scale, as in reference).
// 8 waves/block, each wave owns 16 q rows; k tiles of 32.
__global__ void __launch_bounds__(256)
flash_attn_wmma(const unsigned short* __restrict__ Q,
                const unsigned short* __restrict__ Kt,
                const unsigned short* __restrict__ V,
                unsigned short* __restrict__ Ctx) {
  __shared__ alignas(16) unsigned short pst[8][16][32];  // per-wave P staging
  const int lane = threadIdx.x & 31, wave = threadIdx.x >> 5;
  const int lm = lane & 15, half = lane >> 4;
  int idx = blockIdx.x;
  const int qblk = idx & 15;  idx >>= 4;   // kS/128 == 16
  const int h = idx & 15;                  // kH == 16
  const int b = idx >> 4;
  const int q0 = qblk * 128 + wave * 16;
  const float slope = exp2f(-0.5f * (float)(h + 1));   // 2^(-8(h+1)/H)
  const float rsc = 0.088388347648318447f;             // 1/sqrt(128)

  // Q fragments: 16 rows x 128 d = 4 A-frags (persistent)
  AF qf[4];
  const unsigned short* qp =
      Q + (size_t)(b * kS + q0 + lm) * kE + h * kD + half * 8;
#pragma unroll
  for (int c = 0; c < 4; ++c) {
    qf[c].q[0] = *(const uint4*)(qp + c * 32);
    qf[c].q[1] = *(const uint4*)(qp + c * 32 + 16);
  }

  float mrow[8], lrow[8];
  v8f acc[8] = {};  // ctx 16x128 fp32
#pragma unroll
  for (int e = 0; e < 8; ++e) { mrow[e] = -3e38f; lrow[e] = 0.f; }

  const unsigned short* ktb = Kt + (size_t)(b * kH + h) * kD * kS;
  const unsigned short* vb = V + (size_t)b * kS * kE + h * kD;

  for (int k0 = 0; k0 < q0 + 16; k0 += 32) {
    // scores: two 16x16 tiles, chained over d (4 x k=32 each)
    v8f s[2] = {};
#pragma unroll
    for (int t = 0; t < 2; ++t)
#pragma unroll
      for (int c = 0; c < 4; ++c) {
        AF kf;  // B-frag col k0+t*16+lm of Kt, d rows c*32 + table-order k
        const unsigned short* kp =
            ktb + (size_t)(c * 32) * kS + k0 + t * 16 + lm;
        unsigned short* ku = (unsigned short*)&kf;
#pragma unroll
        for (int s = 0; s < 16; ++s)
          ku[s] = kp[(size_t)((s & 7) + half * 8 + (s >> 3) * 16) * kS];
        s[t] = wmma_bf16(qf[c].v, kf.v, s[t]);
      }

    // mask -> +alibi -> *1/sqrt(D); online softmax (16-lane row reduce)
    float p0[8], p1[8];
#pragma unroll
    for (int e = 0; e < 8; ++e) {
      const int qr = q0 + half * 8 + e;
      const int g0 = k0 + lm, g1 = k0 + 16 + lm;
      float v0 = s[0][e], v1 = s[1][e];
      if (g0 > qr) v0 = kNeg;
      if (g1 > qr) v1 = kNeg;
      v0 = (v0 + slope * (float)(g0 - qr)) * rsc;
      v1 = (v1 + slope * (float)(g1 - qr)) * rsc;
      float mt = fmaxf(v0, v1);
#pragma unroll
      for (int off = 1; off < 16; off <<= 1)
        mt = fmaxf(mt, __shfl_xor(mt, off, 32));
      const float mn = fmaxf(mrow[e], mt);
      const float sc = __expf(mrow[e] - mn);
      mrow[e] = mn;
      const float e0 = __expf(v0 - mn), e1 = __expf(v1 - mn);
      p0[e] = e0;
      p1[e] = e1;
      float ps = e0 + e1;
#pragma unroll
      for (int off = 1; off < 16; off <<= 1) ps += __shfl_xor(ps, off, 32);
      lrow[e] = lrow[e] * sc + ps;
#pragma unroll
      for (int j = 0; j < 8; ++j) acc[j][e] *= sc;
    }

    // restage P (C-layout fp32) -> LDS -> A-layout bf16 frag (wave-private)
#pragma unroll
    for (int e = 0; e < 8; ++e) {
      pst[wave][half * 8 + e][lm] = f2bf(p0[e]);
      pst[wave][half * 8 + e][16 + lm] = f2bf(p1[e]);
    }
    __builtin_amdgcn_wave_barrier();
    asm volatile("s_wait_dscnt 0x0" ::: "memory");
    AF pf;
    pf.q[0] = *(const uint4*)&pst[wave][lm][half * 8];
    pf.q[1] = *(const uint4*)&pst[wave][lm][16 + half * 8];
    __builtin_amdgcn_wave_barrier();

    // ctx += P(16x32) @ V(32x128): 8 WMMAs
#pragma unroll
    for (int c2 = 0; c2 < 8; ++c2) {
      AF vf;
      const unsigned short* vp = vb + (size_t)k0 * kE + c2 * 16 + lm;
      unsigned short* vu = (unsigned short*)&vf;
#pragma unroll
      for (int s = 0; s < 16; ++s)
        vu[s] = vp[(size_t)((s & 7) + half * 8 + (s >> 3) * 16) * kE];
      acc[c2] = wmma_bf16(pf.v, vf.v, acc[c2]);
    }
  }

  // normalize and store ctx (bf16) for the output projection GEMM
  unsigned short* cp = Ctx + (size_t)b * kS * kE + h * kD;
#pragma unroll
  for (int e = 0; e < 8; ++e) {
    const float inv = 1.f / lrow[e];
    const int row = q0 + half * 8 + e;
#pragma unroll
    for (int c2 = 0; c2 < 8; ++c2)
      cp[(size_t)row * kE + c2 * 16 + lm] = f2bf(acc[c2][e] * inv);
  }
}

}  // namespace

extern "C" void kernel_launch(void* const* d_in, const int* in_sizes, int n_in,
                              void* d_out, int out_size, void* d_ws,
                              size_t ws_size, hipStream_t stream) {
  (void)in_sizes; (void)n_in; (void)out_size; (void)ws_size;
  const float* x  = (const float*)d_in[0];
  const float* Wq = (const float*)d_in[1];
  const float* Wk = (const float*)d_in[2];
  const float* Wv = (const float*)d_in[3];
  const float* Wo = (const float*)d_in[4];
  const float* bo = (const float*)d_in[5];
  float* out = (float*)d_out;

  char* p = (char*)d_ws;
  const size_t actB = (size_t)kM * kE * 2;  // 16.78 MB (bf16 activations)
  const size_t wB = (size_t)kE * kE * 2;    //  8.39 MB (bf16 weights)
  unsigned short* xb  = (unsigned short*)p; p += actB;
  unsigned short* wqb = (unsigned short*)p; p += wB;
  unsigned short* wkb = (unsigned short*)p; p += wB;
  unsigned short* wvb = (unsigned short*)p; p += wB;
  unsigned short* wob = (unsigned short*)p; p += wB;
  unsigned short* qb  = (unsigned short*)p; p += actB;
  unsigned short* kb  = (unsigned short*)p; p += actB;
  unsigned short* vbp = (unsigned short*)p; p += actB;
  unsigned short* ktb = (unsigned short*)p; p += actB;
  unsigned short* ctb = (unsigned short*)p; p += actB;  // total ~134 MB

  cast_f32_to_bf16<<<kM * kE / 8 / 256, 256, 0, stream>>>(x, xb, kM * kE);
  cast_f32_to_bf16<<<kE * kE / 8 / 256, 256, 0, stream>>>(Wq, wqb, kE * kE);
  cast_f32_to_bf16<<<kE * kE / 8 / 256, 256, 0, stream>>>(Wk, wkb, kE * kE);
  cast_f32_to_bf16<<<kE * kE / 8 / 256, 256, 0, stream>>>(Wv, wvb, kE * kE);
  cast_f32_to_bf16<<<kE * kE / 8 / 256, 256, 0, stream>>>(Wo, wob, kE * kE);

  dim3 gg(kE / 256, kM / 128);
  gemm_bf16_wmma<false><<<gg, 256, 0, stream>>>(xb, wqb, qb, nullptr, nullptr,
                                                kM, kE, kE);
  gemm_bf16_wmma<false><<<gg, 256, 0, stream>>>(xb, wkb, kb, nullptr, nullptr,
                                                kM, kE, kE);
  gemm_bf16_wmma<false><<<gg, 256, 0, stream>>>(xb, wvb, vbp, nullptr, nullptr,
                                                kM, kE, kE);

  transpose_k<<<dim3(kS / 32, kD / 32, kB * kH), dim3(32, 8), 0, stream>>>(kb,
                                                                           ktb);

  flash_attn_wmma<<<kB * kH * (kS / 128), 256, 0, stream>>>(qb, ktb, vbp, ctb);

  gemm_bf16_wmma<true><<<gg, 256, 0, stream>>>(ctb, wob, nullptr, out, bo, kM,
                                               kE, kE);
}
